// AutoEncoder_48052094108202
// MI455X (gfx1250) — matrix operand order repaired and checked by reading, not compiled, not run on hardware
//
#include <hip/hip_runtime.h>
#include <hip/hip_bf16.h>
#include <math.h>

// ---------------------------------------------------------------------------
// Types for CDNA5 WMMA (wave32): bf16 16x16x32 -> f32 accumulators
// ---------------------------------------------------------------------------
typedef __bf16 bf16_t;
typedef __attribute__((ext_vector_type(16))) __bf16    v16bf;
typedef __attribute__((ext_vector_type(8)))  float     v8f;
typedef __attribute__((ext_vector_type(8)))  unsigned  v8u;

#define TILE_M 64
#define TILE_N 128
#define TILE_K 32
#define AS_STRIDE 34   // padded bf16 stride (k-dim) -> 17 dwords, conflict-free
#define BS_STRIDE 34   // B stored transposed [n][k], same padding

__device__ __forceinline__ float bfh(float x) { return (float)(bf16_t)x; }

// pack two floats as two bf16 (RNE) into one dword (little-endian: a -> low)
__device__ __forceinline__ unsigned pack2(float a, float b) {
    const unsigned short ua = __builtin_bit_cast(unsigned short, (bf16_t)a);
    const unsigned short ub = __builtin_bit_cast(unsigned short, (bf16_t)b);
    return (unsigned)ua | ((unsigned)ub << 16);
}

// ---------------------------------------------------------------------------
// C[M,N] = A[M,K] @ B[K,N] + bias[N], fp32 in/out.
// Split-precision bf16 (hi+lo), 3x V_WMMA_F32_16X16X32_BF16 per subtile.
// Block tile 64x128, 8 wave32s in a 2x4 grid, each wave computes 32x32.
// LDS ping-pong double buffering; global loads for tile t+1 issued before
// the WMMA block of tile t; one barrier per K-iteration.
// ---------------------------------------------------------------------------
__global__ __launch_bounds__(256, 2)
void gemm_bias_bf16x3(const float* __restrict__ A,
                      const float* __restrict__ Bm,
                      const float* __restrict__ bias,
                      float* __restrict__ C,
                      int M, int N, int K)
{
    __shared__ bf16_t As_hi[2][TILE_M * AS_STRIDE];
    __shared__ bf16_t As_lo[2][TILE_M * AS_STRIDE];
    __shared__ bf16_t Bs_hi[2][TILE_N * BS_STRIDE];
    __shared__ bf16_t Bs_lo[2][TILE_N * BS_STRIDE];

    const int tid     = threadIdx.x;
    const int lane    = tid & 31;
    const int wid     = tid >> 5;
    const int wm      = wid >> 2;      // 0..1  (M direction)
    const int wn      = wid & 3;       // 0..3  (N direction)
    const int lane_lo = lane & 15;
    const int hlf     = lane >> 4;     // half-wave select

    const int m0 = blockIdx.y * TILE_M;
    const int n0 = blockIdx.x * TILE_N;

    // --- per-thread staging ownership ------------------------------------
    // A tile 64x32: chunk c = tid + i*256 -> row = c>>3, col = (c&7)*4
    const int a_row0 = tid >> 3;
    const int a_row1 = (tid + 256) >> 3;
    const int a_col  = (tid & 7) << 2;
    // B tile 32x128: chunk c = tid + i*256 -> n4 = (c&31)*4, k2 = (c>>5)*2
    // (each thread owns rows k2,k2+1 of one 4-column chunk -> packed stores)
    const int b_n4  = (tid & 31) << 2;
    const int b_k20 = (tid >> 5) << 1;
    const int b_k21 = b_k20 + 16;

    float4 ra[2], rb[2][2];

    auto g_load = [&](int kt) {
        ra[0]    = *(const float4*)(A  + (size_t)(m0 + a_row0) * K + kt + a_col);
        ra[1]    = *(const float4*)(A  + (size_t)(m0 + a_row1) * K + kt + a_col);
        rb[0][0] = *(const float4*)(Bm + (size_t)(kt + b_k20    ) * N + n0 + b_n4);
        rb[0][1] = *(const float4*)(Bm + (size_t)(kt + b_k20 + 1) * N + n0 + b_n4);
        rb[1][0] = *(const float4*)(Bm + (size_t)(kt + b_k21    ) * N + n0 + b_n4);
        rb[1][1] = *(const float4*)(Bm + (size_t)(kt + b_k21 + 1) * N + n0 + b_n4);
    };

    auto s_store = [&](int p) {
        // A: row-major [m][k], hi/lo split, packed dword stores
        #pragma unroll
        for (int i = 0; i < 2; ++i) {
            const int row = i ? a_row1 : a_row0;
            const float4 v = ra[i];
            const float h0 = bfh(v.x), h1 = bfh(v.y), h2 = bfh(v.z), h3 = bfh(v.w);
            bf16_t* ah = &As_hi[p][row * AS_STRIDE + a_col];
            bf16_t* al = &As_lo[p][row * AS_STRIDE + a_col];
            *(unsigned*)(ah)     = pack2(v.x, v.y);
            *(unsigned*)(ah + 2) = pack2(v.z, v.w);
            *(unsigned*)(al)     = pack2(v.x - h0, v.y - h1);
            *(unsigned*)(al + 2) = pack2(v.z - h2, v.w - h3);
        }
        // B: transposed [n][k], k-pairs contiguous -> packed dword stores
        #pragma unroll
        for (int i = 0; i < 2; ++i) {
            const int k2 = i ? b_k21 : b_k20;
            const float c0[4] = {rb[i][0].x, rb[i][0].y, rb[i][0].z, rb[i][0].w};
            const float c1[4] = {rb[i][1].x, rb[i][1].y, rb[i][1].z, rb[i][1].w};
            #pragma unroll
            for (int j = 0; j < 4; ++j) {
                const int n = b_n4 + j;
                const float h0 = bfh(c0[j]), h1 = bfh(c1[j]);
                *(unsigned*)&Bs_hi[p][n * BS_STRIDE + k2] = pack2(c0[j], c1[j]);
                *(unsigned*)&Bs_lo[p][n * BS_STRIDE + k2] = pack2(c0[j] - h0, c1[j] - h1);
            }
        }
    };

    v8f acc[2][2];
    #pragma unroll
    for (int mi = 0; mi < 2; ++mi)
        #pragma unroll
        for (int ni = 0; ni < 2; ++ni)
            acc[mi][ni] = (v8f){0.f,0.f,0.f,0.f,0.f,0.f,0.f,0.f};

    // prologue: fill buffer 0
    g_load(0);
    s_store(0);
    __syncthreads();

    int p = 0;
    for (int kt = 0; kt < K; kt += TILE_K) {
        const bool has_next = (kt + TILE_K) < K;
        if (has_next) g_load(kt + TILE_K);   // issue next tile's loads early

        // ---- load WMMA fragments from LDS buffer p -----------------------
        // A 16x32 bf16: VGPR v<4: K = hlf*8+2v; v>=4: K = 16+hlf*8+2(v-4)
        v16bf ah[2], al[2], bh[2], bl[2];
        #pragma unroll
        for (int mi = 0; mi < 2; ++mi) {
            const int m = wm * 32 + mi * 16 + lane_lo;
            const unsigned* ph = (const unsigned*)(&As_hi[p][m * AS_STRIDE]);
            const unsigned* pl = (const unsigned*)(&As_lo[p][m * AS_STRIDE]);
            v8u uh, ul;
            #pragma unroll
            for (int v = 0; v < 8; ++v) {
                const int kb = (v < 4) ? (hlf * 8 + 2 * v) : (16 + hlf * 8 + 2 * (v - 4));
                uh[v] = ph[kb >> 1];
                ul[v] = pl[kb >> 1];
            }
            ah[mi] = __builtin_bit_cast(v16bf, uh);
            al[mi] = __builtin_bit_cast(v16bf, ul);
        }
        // B 32x16 bf16: lane holds N=lane_lo, K same order as A (v<4: hlf*8+2v; v>=4: 16+hlf*8+2(v-4))
        #pragma unroll
        for (int ni = 0; ni < 2; ++ni) {
            const int n = wn * 32 + ni * 16 + lane_lo;
            const unsigned* ph = (const unsigned*)(&Bs_hi[p][n * BS_STRIDE]);
            const unsigned* pl = (const unsigned*)(&Bs_lo[p][n * BS_STRIDE]);
            v8u uh, ul;
            #pragma unroll
            for (int v = 0; v < 8; ++v) { const int kb = (v < 4) ? (hlf * 8 + 2 * v) : (16 + hlf * 8 + 2 * (v - 4)); uh[v] = ph[kb >> 1]; ul[v] = pl[kb >> 1]; }
            bh[ni] = __builtin_bit_cast(v16bf, uh);
            bl[ni] = __builtin_bit_cast(v16bf, ul);
        }

        // ---- 3-term split-precision accumulation: hi*hi + hi*lo + lo*hi --
        #pragma unroll
        for (int mi = 0; mi < 2; ++mi)
            #pragma unroll
            for (int ni = 0; ni < 2; ++ni) {
                acc[mi][ni] = __builtin_amdgcn_wmma_f32_16x16x32_bf16(
                    false, ah[mi], false, bh[ni], (short)0, acc[mi][ni], false, false);
                acc[mi][ni] = __builtin_amdgcn_wmma_f32_16x16x32_bf16(
                    false, ah[mi], false, bl[ni], (short)0, acc[mi][ni], false, false);
                acc[mi][ni] = __builtin_amdgcn_wmma_f32_16x16x32_bf16(
                    false, al[mi], false, bh[ni], (short)0, acc[mi][ni], false, false);
            }

        if (has_next) s_store(p ^ 1);        // fill the other buffer
        __syncthreads();
        p ^= 1;
    }

    // ---- epilogue: add bias, store C -------------------------------------
    // C/D 16x16 f32 layout: VGPR r -> M = hlf*8 + r, N = lane_lo
    #pragma unroll
    for (int mi = 0; mi < 2; ++mi)
        #pragma unroll
        for (int ni = 0; ni < 2; ++ni) {
            const int col_g = n0 + wn * 32 + ni * 16 + lane_lo;
            const float bv = bias[col_g];
            #pragma unroll
            for (int r = 0; r < 8; ++r) {
                const int row_g = m0 + wm * 32 + mi * 16 + hlf * 8 + r;
                C[(size_t)row_g * N + col_g] = acc[mi][ni][r] + bv;
            }
        }
}

// ---------------------------------------------------------------------------
// seg_start[s] = first index j with segment_ids[j] >= s  (ids are sorted)
// ---------------------------------------------------------------------------
__global__ void seg_bounds_kernel(const int* __restrict__ seg_ids,
                                  int* __restrict__ seg_start, int D, int S)
{
    const int s = blockIdx.x * blockDim.x + threadIdx.x;
    if (s > S) return;
    int lo = 0, hi = D;
    while (lo < hi) {
        const int mid = (lo + hi) >> 1;
        if (seg_ids[mid] < s) lo = mid + 1; else hi = mid;
    }
    seg_start[s] = lo;
}

// ---------------------------------------------------------------------------
// In-place segmented softmax: one block per row; each thread exclusively owns
// segments (disjoint contiguous spans -> no atomics needed).
// ---------------------------------------------------------------------------
__global__ __launch_bounds__(256)
void segsoftmax_kernel(float* __restrict__ Y,
                       const int* __restrict__ seg_start, int D, int S)
{
    __shared__ float row[4096];
    const int r = blockIdx.x;
    float* y = Y + (size_t)r * D;

    for (int j = threadIdx.x * 4; j < D; j += 256 * 4)
        *(float4*)(row + j) = *(const float4*)(y + j);
    __syncthreads();

    for (int s = threadIdx.x; s < S; s += 256) {
        const int beg = seg_start[s];
        const int end = seg_start[s + 1];
        if (beg >= end) continue;
        float m = -INFINITY;
        for (int j = beg; j < end; ++j) m = fmaxf(m, row[j]);
        float sum = 0.0f;
        for (int j = beg; j < end; ++j) {
            const float e = __expf(row[j] - m);
            row[j] = e;
            sum += e;
        }
        const float inv = 1.0f / sum;
        for (int j = beg; j < end; ++j) y[j] = row[j] * inv;
    }
}

// ---------------------------------------------------------------------------
// Launch: seg bounds -> GEMM1 (x@W1+b1 -> h in ws) -> GEMM2 (h@W2+b2 -> out)
//         -> in-place segmented softmax on out.
// ---------------------------------------------------------------------------
extern "C" void kernel_launch(void* const* d_in, const int* in_sizes, int n_in,
                              void* d_out, int out_size, void* d_ws, size_t ws_size,
                              hipStream_t stream)
{
    const float* x   = (const float*)d_in[0];
    const int*   seg = (const int*)  d_in[1];
    const float* W1  = (const float*)d_in[2];
    const float* b1  = (const float*)d_in[3];
    const float* W2  = (const float*)d_in[4];
    const float* b2  = (const float*)d_in[5];
    float*       out = (float*)d_out;

    const int D = in_sizes[1];            // 4096
    const int H = in_sizes[3];            // 2048
    const int B = in_sizes[0] / D;        // 8192
    const int S = 512;                    // NUM_SEGMENTS (static in reference)

    int*   seg_start = (int*)d_ws;                        // (S+1) ints
    float* h         = (float*)((char*)d_ws + 4096);      // B*H fp32 (64 MB)

    seg_bounds_kernel<<<(S + 256) / 256, 256, 0, stream>>>(seg, seg_start, D, S);

    gemm_bias_bf16x3<<<dim3(H / TILE_N, B / TILE_M), dim3(256), 0, stream>>>(
        x, W1, b1, h, B, H, D);
    gemm_bias_bf16x3<<<dim3(D / TILE_N, B / TILE_M), dim3(256), 0, stream>>>(
        h, W2, b2, out, B, D, H);

    segsoftmax_kernel<<<B, dim3(256), 0, stream>>>(out, seg_start, D, S);
}
